// SelectMolAttachment_49160195670281
// MI455X (gfx1250) — compile-verified, matrix operand order checked by reading, not run on hardware
//
#include <hip/hip_runtime.h>

typedef __bf16 bf16_t;
typedef __attribute__((ext_vector_type(16))) __bf16 v16bf;
typedef __attribute__((ext_vector_type(8)))  float  v8f;

#define B_MOL   2048
#define D_IN    256
#define H_DIM   512
#define G3      1536   // 3*H
#define MA_DIM  512
#define FEAT    1024   // MA + H
#define MAXA    50

// ---------------------------------------------------------------------------
// bf16 GEMM, register-blocked: each wave32 computes a 64x32 block of C
// (4 M-tiles x 2 N-tiles), v_wmma_f32_16x16x32_bf16 over K in chunks of 32.
// Per 32-K chunk: 6 fragments (12 x global_load_b128) -> 8 WMMAs.
// A row-major (lda); Bt is the weight pre-transposed to (N,K) row-major (ldb).
// C[M,N] = act(A * Bt^T + bias), written fp32 (Cf) and/or bf16 (Cb).
// ---------------------------------------------------------------------------
__global__ __launch_bounds__(128)
void wmma_gemm_bf16(const bf16_t* __restrict__ A, int lda,
                    const bf16_t* __restrict__ Bt, int ldb,
                    const float* __restrict__ bias,
                    float* __restrict__ Cf, bf16_t* __restrict__ Cb,
                    int ldc, int M, int N, int K, int act)
{
    const int wave = threadIdx.x >> 5;                 // 0..3
    const int lane = threadIdx.x & 31;
    const int half = lane >> 4;                        // K-half held by this lane
    const int l16  = lane & 15;
    const int n0 = blockIdx.x * 128 + wave * 32;       // 2 N-tiles: n0, n0+16
    const int m0 = blockIdx.y * 64;                    // 4 M-tiles
    if (n0 >= N || m0 >= M) return;                    // wave-uniform

    const bf16_t* Ar[4];
    const bf16_t* Br[2];
#pragma unroll
    for (int mi = 0; mi < 4; ++mi) Ar[mi] = A  + (size_t)(m0 + 16 * mi + l16) * lda;
#pragma unroll
    for (int ni = 0; ni < 2; ++ni) Br[ni] = Bt + (size_t)(n0 + 16 * ni + l16) * ldb;

    v8f acc[4][2] = {};
    for (int k0 = 0; k0 < K; k0 += 32) {
        v16bf af[4], bfr[2];
#pragma unroll
        for (int j = 0; j < 8; ++j) {
            // 16-bit fragment layout: VGPR j<4 -> K = half*8 + 2j(+1),
            // VGPR j>=4 -> K = 16 + half*8 + 2(j-4)(+1)  (pairs contiguous,
            // each half-fragment is one 16B chunk -> global_load_b128)
            const int kb = k0 + ((j < 4) ? (half * 8 + 2 * j)
                                         : (16 + half * 8 + 2 * (j - 4)));
#pragma unroll
            for (int mi = 0; mi < 4; ++mi) {
                af[mi][2 * j]     = Ar[mi][kb];
                af[mi][2 * j + 1] = Ar[mi][kb + 1];
            }
#pragma unroll
            for (int ni = 0; ni < 2; ++ni) {
                bfr[ni][2 * j]     = Br[ni][kb];
                bfr[ni][2 * j + 1] = Br[ni][kb + 1];
            }
        }
#pragma unroll
        for (int mi = 0; mi < 4; ++mi)
#pragma unroll
            for (int ni = 0; ni < 2; ++ni)
                acc[mi][ni] = __builtin_amdgcn_wmma_f32_16x16x32_bf16(
                    false, af[mi], false, bfr[ni], (short)0, acc[mi][ni],
                    false, false);
    }

    // C/D layout: lane half h, VGPR v -> M = v + 8*h, N = l16
#pragma unroll
    for (int ni = 0; ni < 2; ++ni) {
        const int n = n0 + 16 * ni + l16;
        const float bv = bias ? bias[n] : 0.0f;
#pragma unroll
        for (int mi = 0; mi < 4; ++mi) {
            const int mb = m0 + 16 * mi + 8 * half;
#pragma unroll
            for (int v = 0; v < 8; ++v) {
                float val = acc[mi][ni][v] + bv;
                if (act) val = fmaxf(val, 0.0f);
                const size_t o = (size_t)(mb + v) * ldc + n;
                if (Cf) Cf[o] = val;
                if (Cb) Cb[o] = (bf16_t)val;
            }
        }
    }
}

// ---------------------------------------------------------------------------
// One-time prep kernels
// ---------------------------------------------------------------------------
__global__ void transpose_bf16_kernel(const float* __restrict__ W,
                                      bf16_t* __restrict__ Wt, int R, int C)
{
    int i = blockIdx.x * blockDim.x + threadIdx.x;
    if (i >= R * C) return;
    int r = i / C, c = i - r * C;
    Wt[(size_t)c * R + r] = (bf16_t)W[i];
}

__global__ void init_state_kernel(const float* __restrict__ molA,
                                  float* __restrict__ h,
                                  bf16_t* __restrict__ featbuf)
{
    int i = blockIdx.x * blockDim.x + threadIdx.x;   // over B*H
    int b = i >> 9, j = i & 511;
    h[i] = 0.0f;
    featbuf[(size_t)b * FEAT + j]       = (bf16_t)molA[(size_t)b * MA_DIM + j];
    featbuf[(size_t)b * FEAT + 512 + j] = (bf16_t)0.0f;
}

__global__ void zero_out_kernel(float* __restrict__ out, int n)
{
    int i = blockIdx.x * blockDim.x + threadIdx.x;
    if (i < n) out[i] = 0.0f;
}

// ---------------------------------------------------------------------------
// Per-step kernels
// ---------------------------------------------------------------------------
__global__ void gather_x_kernel(const float* __restrict__ node_hiddens,
                                const int* __restrict__ offsets,
                                bf16_t* __restrict__ xbf, int t, int Ntot)
{
    int i = blockIdx.x * blockDim.x + threadIdx.x;   // over B*D
    int b = i >> 8, j = i & 255;
    long idx = (long)offsets[b] + t;                 // safe_idx = clip
    if (idx < 0) idx = 0;
    if (idx >= Ntot) idx = Ntot - 1;
    xbf[i] = (bf16_t)node_hiddens[idx * D_IN + j];
}

__global__ void gru_gate_kernel(const float* __restrict__ GI,
                                const float* __restrict__ GH,
                                float* __restrict__ h,
                                bf16_t* __restrict__ featbuf,
                                const int* __restrict__ counts, int t)
{
    int i = blockIdx.x * blockDim.x + threadIdx.x;   // over B*H
    int b = i >> 9, j = i & 511;
    const float* gi = GI + (size_t)b * G3;
    const float* gh = GH + (size_t)b * G3;
    float i_r = gi[j], i_z = gi[512 + j], i_n = gi[1024 + j];
    float h_r = gh[j], h_z = gh[512 + j], h_n = gh[1024 + j];
    float r = 1.0f / (1.0f + __expf(-(i_r + h_r)));
    float z = 1.0f / (1.0f + __expf(-(i_z + h_z)));
    float n = tanhf(i_n + r * h_n);
    float hold = h[i];
    float hnew = (1.0f - z) * n + z * hold;
    float h2 = (t < counts[b]) ? hnew : hold;        // inactive graphs keep h
    h[i] = h2;
    featbuf[(size_t)b * FEAT + 512 + j] = (bf16_t)h2;
}

__global__ void head_kernel(const float* __restrict__ Y2,
                            const float* __restrict__ W3,
                            const float* __restrict__ b3,
                            const int* __restrict__ counts,
                            const int* __restrict__ offsets,
                            float* __restrict__ out, int t)
{
    int wave = threadIdx.x >> 5, lane = threadIdx.x & 31;
    int b = blockIdx.x * 4 + wave;
    const float* y = Y2 + (size_t)b * 256;
    float acc = 0.0f;
#pragma unroll
    for (int e = 0; e < 8; ++e) {
        int k = lane + 32 * e;
        acc += y[k] * W3[k];
    }
#pragma unroll
    for (int off = 16; off; off >>= 1) acc += __shfl_xor(acc, off, 32);
    if (lane == 0 && t < counts[b]) {
        float v = 1.0f / (1.0f + __expf(-(acc + b3[0])));
        out[offsets[b] + t] = v;
    }
}

// ---------------------------------------------------------------------------
extern "C" void kernel_launch(void* const* d_in, const int* in_sizes, int n_in,
                              void* d_out, int out_size, void* d_ws, size_t ws_size,
                              hipStream_t stream)
{
    const float* molA   = (const float*)d_in[0];
    const float* nodeH  = (const float*)d_in[1];
    const float* W_ih   = (const float*)d_in[2];
    const float* W_hh   = (const float*)d_in[3];
    const float* b_ih   = (const float*)d_in[4];
    const float* b_hh   = (const float*)d_in[5];
    const float* W1     = (const float*)d_in[6];
    const float* b1     = (const float*)d_in[7];
    const float* W2     = (const float*)d_in[8];
    const float* b2     = (const float*)d_in[9];
    const float* W3     = (const float*)d_in[10];
    const float* b3     = (const float*)d_in[11];
    const int*   counts = (const int*)d_in[12];
    const int*   offs   = (const int*)d_in[13];
    float* out = (float*)d_out;
    const int Ntot = in_sizes[1] / D_IN;

    // workspace carve-up (256B aligned)
    char* wsp = (char*)d_ws;
    auto alloc = [&](size_t bytes) {
        char* p = wsp;
        wsp += (bytes + 255) & ~(size_t)255;
        return p;
    };
    bf16_t* WihT = (bf16_t*)alloc((size_t)G3 * D_IN * 2);     // (1536,256)
    bf16_t* WhhT = (bf16_t*)alloc((size_t)G3 * H_DIM * 2);    // (1536,512)
    bf16_t* W1T  = (bf16_t*)alloc((size_t)512 * FEAT * 2);    // (512,1024)
    bf16_t* W2T  = (bf16_t*)alloc((size_t)256 * 512 * 2);     // (256,512)
    bf16_t* feat = (bf16_t*)alloc((size_t)B_MOL * FEAT * 2);  // [molA | h] bf16
    bf16_t* xbf  = (bf16_t*)alloc((size_t)B_MOL * D_IN * 2);
    float*  h    = (float*) alloc((size_t)B_MOL * H_DIM * 4);
    float*  GI   = (float*) alloc((size_t)B_MOL * G3 * 4);
    float*  GH   = (float*) alloc((size_t)B_MOL * G3 * 4);
    bf16_t* Y1   = (bf16_t*)alloc((size_t)B_MOL * 512 * 2);
    float*  Y2   = (float*) alloc((size_t)B_MOL * 256 * 4);

    auto gemm = [&](const bf16_t* A, int lda, const bf16_t* Bt, int ldb,
                    const float* bias, float* Cf, bf16_t* Cb,
                    int M, int N, int K, int act) {
        dim3 grid(N / 128, M / 64);
        wmma_gemm_bf16<<<grid, 128, 0, stream>>>(A, lda, Bt, ldb, bias,
                                                 Cf, Cb, N, M, N, K, act);
    };

    // one-time prep
    transpose_bf16_kernel<<<(D_IN * G3 + 255) / 256, 256, 0, stream>>>(W_ih, WihT, D_IN, G3);
    transpose_bf16_kernel<<<(H_DIM * G3 + 255) / 256, 256, 0, stream>>>(W_hh, WhhT, H_DIM, G3);
    transpose_bf16_kernel<<<(FEAT * 512 + 255) / 256, 256, 0, stream>>>(W1, W1T, FEAT, 512);
    transpose_bf16_kernel<<<(512 * 256 + 255) / 256, 256, 0, stream>>>(W2, W2T, 512, 256);
    init_state_kernel<<<(B_MOL * H_DIM) / 256, 256, 0, stream>>>(molA, h, feat);
    zero_out_kernel<<<(Ntot + 255) / 256, 256, 0, stream>>>(out, Ntot);

    // 3 warm-up passes + 1 output pass, hidden carried throughout
    for (int pass = 0; pass < 4; ++pass) {
        const bool final_pass = (pass == 3);
        for (int t = 0; t < MAXA; ++t) {
            gather_x_kernel<<<(B_MOL * D_IN) / 256, 256, 0, stream>>>(nodeH, offs, xbf, t, Ntot);
            // gi = x @ W_ih + b_ih          (2048x1536, K=256)
            gemm(xbf, D_IN, WihT, D_IN, b_ih, GI, nullptr, B_MOL, G3, D_IN, 0);
            // gh = h @ W_hh + b_hh          (2048x1536, K=512)
            gemm(feat + 512, FEAT, WhhT, H_DIM, b_hh, GH, nullptr, B_MOL, G3, H_DIM, 0);
            gru_gate_kernel<<<(B_MOL * H_DIM) / 256, 256, 0, stream>>>(GI, GH, h, feat, counts, t);
            if (final_pass) {
                // y1 = relu([molA|h] @ W1 + b1)   (2048x512, K=1024)
                gemm(feat, FEAT, W1T, FEAT, b1, nullptr, Y1, B_MOL, 512, FEAT, 1);
                // y2 = relu(y1 @ W2 + b2)         (2048x256, K=512)
                gemm(Y1, 512, W2T, 512, b2, Y2, nullptr, B_MOL, 256, 512, 1);
                head_kernel<<<B_MOL / 4, 128, 0, stream>>>(Y2, W3, b3, counts, offs, out, t);
            }
        }
    }
}
